// MooreVanillaRNN_17549236372147
// MI455X (gfx1250) — compile-verified, matrix operand order checked by reading, not run on hardware
//
#include <hip/hip_runtime.h>
#include <hip/hip_bf16.h>

typedef __attribute__((ext_vector_type(16))) _Float16 v16h;
typedef __attribute__((ext_vector_type(8)))  _Float16 v8h;
typedef __attribute__((ext_vector_type(8)))  float    v8f;

#define RNN_B 128
#define RNN_T 512
#define RNN_D 256
#define RNN_S 512

// Load a 16x32 f16 WMMA fragment (A-layout) from a row-major matrix with
// leading dimension 256 halves. Per ISA 7.12.2:
//   lanes 0-15 : row = lane,    K = kbase+[0..7]  and kbase+16+[0..7]
//   lanes 16-31: row = lane-16, K = kbase+8+[0..7] and kbase+24+[0..7]
// p must point at (row_base + lane&15)*256 + kbase + (lane>=16)*8.
__device__ inline v16h ld_frag(const _Float16* p) {
  v8h lo = *(const v8h*)(p);
  v8h hi = *(const v8h*)(p + 16);
  v16h r;
#pragma unroll
  for (int i = 0; i < 8; ++i) { r[i] = lo[i]; r[i + 8] = hi[i]; }
  return r;
}

__device__ inline v8f wmma_f16(v16h a, v16h b, v8f c) {
  return __builtin_amdgcn_wmma_f32_16x16x32_f16(false, a, false, b, (short)0,
                                                c, false, false);
}

// Branch-free tanh: hardware v_tanh_f32 when available, else exp2/rcp form.
__device__ inline float fast_tanh(float x) {
#if __has_builtin(__builtin_amdgcn_tanhf)
  return __builtin_amdgcn_tanhf(x);
#else
  const float xc = fminf(fmaxf(x, -10.0f), 10.0f);
  const float e  = __builtin_amdgcn_exp2f(xc * 2.8853900817779268f);  // e^(2x)
  return (e - 1.0f) * __builtin_amdgcn_rcpf(e + 1.0f);
#endif
}

// ---------------------------------------------------------------- small prep
__global__ void f32_to_f16_kernel(const float* __restrict__ src,
                                  _Float16* __restrict__ dst, int n) {
  int i = blockIdx.x * 256 + threadIdx.x;
  if (i < n) dst[i] = (_Float16)src[i];
}

__global__ void bias_sum_kernel(const float* __restrict__ a,
                                const float* __restrict__ b,
                                float* __restrict__ out) {
  int i = threadIdx.x;  // D == 256 == blockDim
  out[i] = a[i] + b[i];
}

// gather: x[row, d] = (f16) emb[ids[row], d] ; one block per (b,t) row
__global__ void embed_kernel(const int* __restrict__ ids,
                             const float* __restrict__ emb,
                             _Float16* __restrict__ x) {
  int row = blockIdx.x;
  int d   = threadIdx.x;
  int id  = ids[row];
  x[(size_t)row * RNN_D + d] = (_Float16)emb[(size_t)id * RNN_D + d];
}

// ------------------------------------------------------- batched WMMA GEMM
// C[M,N] = A[M,256] * W[N,256]^T (+ bias[n]); A,W f16 row-major, C = OutT.
// Block: 256 thr = 8 waves; wave (wm,wn) computes rows m0..m0+15, cols n0..n0+63.
// Block tile 32 x 256. Grid: (N/256, M/32). Exact fit -> EXEC all ones.
// K-loop fragments double-buffered in registers to overlap loads with WMMAs.
template <typename OutT>
__global__ void gemm_rnT_kernel(const _Float16* __restrict__ A,
                                const _Float16* __restrict__ W,
                                const float* __restrict__ bias,
                                OutT* __restrict__ C, int N) {
  const int lane = threadIdx.x & 31;
  const int wv   = threadIdx.x >> 5;
  const int wm   = wv >> 2, wn = wv & 3;
  const int lr   = lane & 15;
  const int sel  = lane >> 4;
  const size_t m0 = (size_t)blockIdx.y * 32 + (size_t)wm * 16;
  const int    n0 = blockIdx.x * 256 + wn * 64;

  const _Float16* arow = A + (m0 + lr) * RNN_D + sel * 8;
  const _Float16* wrow = W + (size_t)(n0 + lr) * RNN_D + sel * 8;

  v8f acc[4] = {};
  v16h a = ld_frag(arow);
  v16h b[4];
#pragma unroll
  for (int j = 0; j < 4; ++j) b[j] = ld_frag(wrow + (size_t)j * 16 * RNN_D);

#pragma unroll
  for (int kb = 0; kb < 8; ++kb) {
    v16h an = a;
    v16h bn[4] = {b[0], b[1], b[2], b[3]};
    if (kb < 7) {  // prefetch next k-block while current WMMAs run
      an = ld_frag(arow + (kb + 1) * 32);
#pragma unroll
      for (int j = 0; j < 4; ++j)
        bn[j] = ld_frag(wrow + (size_t)j * 16 * RNN_D + (kb + 1) * 32);
    }
#pragma unroll
    for (int j = 0; j < 4; ++j) acc[j] = wmma_f16(a, b[j], acc[j]);
    a = an;
#pragma unroll
    for (int j = 0; j < 4; ++j) b[j] = bn[j];
  }

#pragma unroll
  for (int j = 0; j < 4; ++j) {
    const int col = n0 + j * 16 + lr;
    const float bv = bias ? bias[col] : 0.0f;
    OutT* cp = C + (m0 + sel * 8) * (size_t)N + col;
#pragma unroll
    for (int r = 0; r < 8; ++r) cp[(size_t)r * N] = (OutT)(acc[j][r] + bv);
  }
}

// ------------------------------------------------------------ recurrent scan
// h_t = tanh(pre[:,t,:] + h_{t-1} @ W_hh^T), per 16-row batch tile.
// Grid: 8 blocks (B/16). Block: 256 thr = 8 waves; wave w owns cols [32w,32w+32).
// __launch_bounds__(256,1): only 8 waves/WGP -> ~512 VGPRs/wave available, so
// the 16 persistent W_hh B-fragments (128 VGPRs) stay register-resident.
// h state double-buffered in LDS -> 1 barrier/step; A-fragments are
// double-buffered in registers so ds_load latency overlaps the WMMA chain;
// h_t is stored to global cooperatively with coalesced b128 stores.
__global__ void __launch_bounds__(256, 1)
rnn_scan_kernel(const _Float16* __restrict__ pre,
                const _Float16* __restrict__ Whh,
                _Float16* __restrict__ h_out) {
  __shared__ _Float16 sH[2][16 * RNN_D];  // 2 x 8 KB

  const int lane  = threadIdx.x & 31;
  const int wv    = threadIdx.x >> 5;
  const int lr    = lane & 15;
  const int sel   = lane >> 4;
  const int bbase = blockIdx.x * 16;
  const int nt0   = 2 * wv;

  // Persistent register-resident W_hh column tiles for this wave.
  v16h bfrag[2][8];
#pragma unroll
  for (int j = 0; j < 2; ++j)
#pragma unroll
    for (int kb = 0; kb < 8; ++kb)
      bfrag[j][kb] = ld_frag(Whh + (size_t)((nt0 + j) * 16 + lr) * RNN_D
                                 + kb * 32 + sel * 8);

  {  // zero h buffer 0
    v8h z = {};
    for (int i = threadIdx.x; i < (16 * RNN_D) / 8; i += 256)
      *(v8h*)(&sH[0][0] + (size_t)i * 8) = z;
  }
  __syncthreads();

  const size_t bt_stride = (size_t)RNN_T * RNN_D;  // stride between batch rows
  const _Float16* prow = pre + (size_t)(bbase + sel * 8) * bt_stride
                             + nt0 * 16 + lr;

  // cooperative-store mapping: thread handles 2 chunks of 8 halves
  const int st_row0 = threadIdx.x >> 4;          // 0..15
  const int st_c0   = (threadIdx.x & 15) * 16;   // 0..240 step 16

  for (int t = 0; t < RNN_T; ++t) {
    const _Float16* sHc = &sH[t & 1][0];
    _Float16*       sHn = &sH[(t + 1) & 1][0];
    const size_t toff = (size_t)t * RNN_D;
    const _Float16* arow = sHc + (size_t)lr * RNN_D + sel * 8;

    // Issue pre loads early; they resolve under the WMMA chain.
    _Float16 pv[2][8];
#pragma unroll
    for (int j = 0; j < 2; ++j)
#pragma unroll
      for (int r = 0; r < 8; ++r)
        pv[j][r] = prow[(size_t)r * bt_stride + toff + j * 16];
    __builtin_prefetch(prow + toff + RNN_D, 0, 3);  // next timestep's row

    v8f acc[2] = {};
    v16h afr = ld_frag(arow);  // k-block 0
#pragma unroll
    for (int kb = 0; kb < 8; ++kb) {
      v16h anx = afr;
      if (kb < 7) anx = ld_frag(arow + (kb + 1) * 32);  // overlap with WMMAs
      acc[0] = wmma_f16(afr, bfrag[0][kb], acc[0]);
      acc[1] = wmma_f16(afr, bfrag[1][kb], acc[1]);
      afr = anx;
    }

#pragma unroll
    for (int j = 0; j < 2; ++j)
#pragma unroll
      for (int r = 0; r < 8; ++r) {
        const _Float16 hv = (_Float16)fast_tanh(acc[j][r] + (float)pv[j][r]);
        sHn[(size_t)(r + sel * 8) * RNN_D + (nt0 + j) * 16 + lr] = hv;
      }
    __syncthreads();  // sHn (h_t) complete

    // Coalesced cooperative store of h_t from LDS (2 x b128 per thread).
    // Safe: step t+1 writes the other buffer; step t+2 (same buffer) cannot
    // start until all threads pass step t+1's barrier.
    _Float16* hg = h_out + (size_t)(bbase + st_row0) * bt_stride + toff + st_c0;
    const _Float16* hs = sHn + (size_t)st_row0 * RNN_D + st_c0;
    *(v8h*)(hg)     = *(const v8h*)(hs);
    *(v8h*)(hg + 8) = *(const v8h*)(hs + 8);
  }
}

// ----------------------------------------------------------------- launcher
extern "C" void kernel_launch(void* const* d_in, const int* in_sizes, int n_in,
                              void* d_out, int out_size, void* d_ws, size_t ws_size,
                              hipStream_t stream) {
  const int*   ids    = (const int*)d_in[0];
  const float* emb    = (const float*)d_in[1];
  const float* W_ih0  = (const float*)d_in[2];
  const float* W_hh0  = (const float*)d_in[3];
  const float* b_ih0  = (const float*)d_in[4];
  const float* b_hh0  = (const float*)d_in[5];
  const float* W_ih1  = (const float*)d_in[6];
  const float* W_hh1  = (const float*)d_in[7];
  const float* b_ih1  = (const float*)d_in[8];
  const float* b_hh1  = (const float*)d_in[9];
  const float* W_head = (const float*)d_in[10];
  float* logits = (float*)d_out;

  char* ws = (char*)d_ws;
  size_t off = 0;
  auto alloc = [&](size_t bytes) -> void* {
    void* p = ws + off;
    off += (bytes + 255) & ~(size_t)255;
    return p;
  };

  const int DD = RNN_D * RNN_D;            // 65536
  const size_t BT = (size_t)RNN_B * RNN_T; // 65536 rows
  _Float16* wih0h  = (_Float16*)alloc(DD * 2);
  _Float16* whh0h  = (_Float16*)alloc(DD * 2);
  _Float16* wih1h  = (_Float16*)alloc(DD * 2);
  _Float16* whh1h  = (_Float16*)alloc(DD * 2);
  _Float16* wheadh = (_Float16*)alloc((size_t)RNN_S * RNN_D * 2);
  float*    bias0  = (float*)alloc(RNN_D * 4);
  float*    bias1  = (float*)alloc(RNN_D * 4);
  _Float16* x_h    = (_Float16*)alloc(BT * RNN_D * 2);  // reused as h2
  _Float16* pre    = (_Float16*)alloc(BT * RNN_D * 2);  // reused per layer
  _Float16* h1_h   = (_Float16*)alloc(BT * RNN_D * 2);
  _Float16* h2_h   = x_h;

  // weights -> f16
  f32_to_f16_kernel<<<DD / 256, 256, 0, stream>>>(W_ih0, wih0h, DD);
  f32_to_f16_kernel<<<DD / 256, 256, 0, stream>>>(W_hh0, whh0h, DD);
  f32_to_f16_kernel<<<DD / 256, 256, 0, stream>>>(W_ih1, wih1h, DD);
  f32_to_f16_kernel<<<DD / 256, 256, 0, stream>>>(W_hh1, whh1h, DD);
  f32_to_f16_kernel<<<(RNN_S * RNN_D) / 256, 256, 0, stream>>>(
      W_head, wheadh, RNN_S * RNN_D);
  bias_sum_kernel<<<1, RNN_D, 0, stream>>>(b_ih0, b_hh0, bias0);
  bias_sum_kernel<<<1, RNN_D, 0, stream>>>(b_ih1, b_hh1, bias1);

  // embedding gather -> f16
  embed_kernel<<<(unsigned)BT, RNN_D, 0, stream>>>(ids, emb, x_h);

  const dim3 gemmBlk(256);
  const dim3 gemmGrdD(RNN_D / 256, (unsigned)(BT / 32));  // N=256
  const dim3 gemmGrdS(RNN_S / 256, (unsigned)(BT / 32));  // N=512

  // layer 0
  gemm_rnT_kernel<_Float16><<<gemmGrdD, gemmBlk, 0, stream>>>(
      x_h, wih0h, bias0, pre, RNN_D);
  rnn_scan_kernel<<<RNN_B / 16, 256, 0, stream>>>(pre, whh0h, h1_h);
  // layer 1
  gemm_rnT_kernel<_Float16><<<gemmGrdD, gemmBlk, 0, stream>>>(
      h1_h, wih1h, bias1, pre, RNN_D);
  rnn_scan_kernel<<<RNN_B / 16, 256, 0, stream>>>(pre, whh1h, h2_h);
  // head (no bias, f32 logits)
  gemm_rnT_kernel<float><<<gemmGrdS, gemmBlk, 0, stream>>>(
      h2_h, wheadh, nullptr, logits, RNN_S);
}
